// Mamba2BlockWrapper_38560216383841
// MI455X (gfx1250) — matrix operand order repaired and checked by reading, not compiled, not run on hardware
//
#include <hip/hip_runtime.h>
#include <hip/hip_bf16.h>
#include <stdint.h>

// ---------------- constants (from reference) ----------------
#define BB      8
#define SS      2048
#define TT      (BB * SS)          // 16384 tokens
#define DMODEL  1024
#define DINNER  2048
#define HEADS   32
#define HDIM    64
#define DSTATE  128
#define CONVD   2304               // DINNER + 2*DSTATE
#define EPROJ   4384               // 2*DINNER + 2*DSTATE + HEADS
#define EPSF    1e-5f

// GEMM tiling
#define BLK_M        128
#define BLK_N        128
#define TILE_K       64
#define LDS_ROW_B    144           // 128B row + 16B TDM pad -> conflict-free ds_load_b128
#define TILE_BYTES   (BLK_M * LDS_ROW_B)   // 18432
#define SMEM_BYTES   (2 * TILE_BYTES)      // 36864 (double buffer)

typedef __attribute__((ext_vector_type(16))) __bf16 v16bf;
typedef __attribute__((ext_vector_type(8)))  float  v8f;
typedef __attribute__((ext_vector_type(4)))  unsigned int u32x4;
typedef __attribute__((ext_vector_type(8)))  int i32x8;
typedef __attribute__((ext_vector_type(4)))  int i32x4;

#if defined(__has_builtin)
#  if __has_builtin(__builtin_amdgcn_tensor_load_to_lds) && __has_builtin(__builtin_amdgcn_s_wait_tensorcnt)
#    define USE_TDM 1
#  else
#    define USE_TDM 0
#  endif
#else
#  define USE_TDM 0
#endif

__device__ __forceinline__ uint16_t f2bf(float f) {
    uint32_t u = __float_as_uint(f);
    uint32_t r = (u + 0x7FFFu + ((u >> 16) & 1u)) >> 16;
    return (uint16_t)r;
}

__device__ __forceinline__ v16bf as_v16bf(uint4 lo, uint4 hi) {
    union { uint4 u[2]; v16bf v; } x;
    x.u[0] = lo; x.u[1] = hi;
    return x.v;
}

#if USE_TDM
// Issue one TDM load of a BLK_M x TILE_K bf16 tile (row-major, row stride K elems)
// into LDS at byte offset ldsoff, with 16B padding after each 128B row.
// 6-arg builtin form: (u32x4 g0, i32x8 g1, i32x4 g2, i32x4 g3, i32x8 g4, i32 cpol)
__device__ __forceinline__ void tdm_load_tile(const uint16_t* gptr, uint32_t ldsoff,
                                              int K, int Mtot) {
    uint64_t ga = (uint64_t)(uintptr_t)gptr;
    u32x4 g0;
    g0[0] = 1u;                                   // count=1, user descriptor
    g0[1] = ldsoff;                               // lds_addr (bytes, within our alloc)
    g0[2] = (uint32_t)ga;                         // global_addr[31:0]
    g0[3] = (uint32_t)((ga >> 32) & 0x1FFFFFFu)   // global_addr[56:32]
          | (2u << 30);                           // type = 2 ("image")
    i32x8 g1;
    g1[0] = (int)(0x00010000u                     // data_size = 1 (2 bytes)
          | (1u << 20)                            // pad_enable
          | (4u << 22)                            // pad_interval: 32 DWORDs (128B)
          | (3u << 25));                          // pad_amount: 4 DWORDs (16B)
    g1[1] = (int)(((uint32_t)K & 0xFFFFu) << 16);               // tensor_dim0[15:0]
    g1[2] = (int)((((uint32_t)K >> 16) & 0xFFFFu)               // tensor_dim0[31:16]
          | (((uint32_t)Mtot & 0xFFFFu) << 16));                // tensor_dim1[15:0]
    g1[3] = (int)((((uint32_t)Mtot >> 16) & 0xFFFFu)            // tensor_dim1[31:16]
          | ((uint32_t)TILE_K << 16));                          // tile_dim0 = 64
    g1[4] = (int)(uint32_t)BLK_M;                               // tile_dim1 = 128
    g1[5] = (int)(uint32_t)K;                                   // tensor_dim0_stride lo
    g1[6] = 0;
    g1[7] = 0;
    i32x4 gz4; gz4[0] = 0; gz4[1] = 0; gz4[2] = 0; gz4[3] = 0;
    i32x8 gz8;
    gz8[0] = 0; gz8[1] = 0; gz8[2] = 0; gz8[3] = 0;
    gz8[4] = 0; gz8[5] = 0; gz8[6] = 0; gz8[7] = 0;
    __builtin_amdgcn_tensor_load_to_lds(g0, g1, gz4, gz4, gz8, 0);
}
#endif

// ---------------- 1) RMSNorm -> bf16 ----------------
__global__ __launch_bounds__(256) void rmsnorm_to_bf16_kernel(
    const float* __restrict__ x, const float* __restrict__ w,
    uint16_t* __restrict__ out)
{
    const int t = blockIdx.x;
    const float* row = x + (size_t)t * DMODEL;
    float ss = 0.f;
    for (int i = threadIdx.x; i < DMODEL; i += 256) { float v = row[i]; ss += v * v; }
    for (int o = 16; o > 0; o >>= 1) ss += __shfl_xor(ss, o, 32);
    __shared__ float red[8];
    __shared__ float scale_s;
    if ((threadIdx.x & 31) == 0) red[threadIdx.x >> 5] = ss;
    __syncthreads();
    if (threadIdx.x == 0) {
        float s = 0.f;
        #pragma unroll
        for (int i = 0; i < 8; ++i) s += red[i];
        scale_s = rsqrtf(s / (float)DMODEL + EPSF);
    }
    __syncthreads();
    const float scale = scale_s;
    for (int i = threadIdx.x; i < DMODEL; i += 256)
        out[(size_t)t * DMODEL + i] = f2bf(row[i] * scale * w[i]);
}

// ---------------- 2) f32 -> bf16 weight conversion ----------------
__global__ __launch_bounds__(256) void f32_to_bf16_kernel(
    const float* __restrict__ in, uint16_t* __restrict__ out, size_t n)
{
    size_t i = (size_t)blockIdx.x * 256 + threadIdx.x;
    size_t stride = (size_t)gridDim.x * 256;
    for (; i < n; i += stride) out[i] = f2bf(in[i]);
}

// ---------------- 3) bf16 WMMA GEMM: C[M,N] = A[M,K] * Bw[N,K]^T (+resid) ----
// Block tile 128x128, 8 waves each computing 32x64 (2x4 WMMA frags).
// A tile staged to LDS via Tensor Data Mover (double-buffered, TENSORcnt-paced);
// B fragments direct from global (weights are L2-resident: 192MB L2).
__global__ __launch_bounds__(256) void gemm_bf16_wmma_kernel(
    const uint16_t* __restrict__ A,   // M x K bf16 row-major
    const uint16_t* __restrict__ Bw,  // N x K bf16 row-major (transposed B)
    const float* __restrict__ resid,  // M x N f32 or unused
    float* __restrict__ C,            // M x N f32
    int M, int N, int K, int use_resid)
{
    extern __shared__ char smem[];    // 2 * TILE_BYTES
    const int lane = threadIdx.x & 31;
    const int wave = threadIdx.x >> 5;
    const int wm = wave & 3;               // 0..3 -> M
    const int wn = wave >> 2;              // 0..1 -> N
    const int mblk = blockIdx.x * BLK_M;
    const int n0 = blockIdx.y * BLK_N + wn * 64;
    const int lm = lane & 15;
    const int lh = lane >> 4;

    v8f zero = {0.f, 0.f, 0.f, 0.f, 0.f, 0.f, 0.f, 0.f};
    v8f acc[2][4];
    #pragma unroll
    for (int i = 0; i < 2; ++i)
        #pragma unroll
        for (int j = 0; j < 4; ++j) acc[i][j] = zero;

    // Clamp B row indices once; OOB columns accumulate garbage but are never stored.
    const uint16_t* brow[4];
    #pragma unroll
    for (int j = 0; j < 4; ++j) {
        int e = n0 + j * 16 + lm;
        if (e > N - 1) e = N - 1;
        brow[j] = Bw + (size_t)e * K;
    }

    const int KT = K >> 6;   // K / TILE_K
    const uint16_t* Ablk = A + (size_t)mblk * K;

#if USE_TDM
    if (threadIdx.x < 32) tdm_load_tile(Ablk, 0u, K, M);
#endif

    for (int kt = 0; kt < KT; ++kt) {
        const int k0 = kt << 6;
        __syncthreads();   // prior compute done before its buffer is overwritten
#if USE_TDM
        if (threadIdx.x < 32) {
            if (kt + 1 < KT) {
                tdm_load_tile(Ablk + (size_t)(k0 + TILE_K),
                              (uint32_t)(((kt + 1) & 1) * TILE_BYTES), K, M);
                __builtin_amdgcn_s_wait_tensorcnt(1);   // tile kt complete
            } else {
                __builtin_amdgcn_s_wait_tensorcnt(0);
            }
        }
#else
        {   // cooperative fallback copy (same LDS layout incl. 16B row pad)
            char* dst = smem + (kt & 1) * TILE_BYTES;
            #pragma unroll
            for (int q = 0; q < 4; ++q) {
                const int c = threadIdx.x * 4 + q;       // 0..1023 16B chunks
                const int row = c >> 3;
                const int col = (c & 7) * 8;             // halfwords
                const uint16_t* gp = Ablk + (size_t)row * K + k0 + col;
                *(uint4*)(dst + row * LDS_ROW_B + col * 2) = *(const uint4*)gp;
            }
        }
#endif
        __syncthreads();
        const char* abase = smem + (kt & 1) * TILE_BYTES;

        #pragma unroll
        for (int kk = 0; kk < 2; ++kk) {
            v16bf a[2], b[4];
            #pragma unroll
            for (int i = 0; i < 2; ++i) {
                const int row = wm * 32 + i * 16 + lm;
                const char* ap = abase + row * LDS_ROW_B + kk * 64 + lh * 16;
                a[i] = as_v16bf(*(const uint4*)ap, *(const uint4*)(ap + 32));
            }
            #pragma unroll
            for (int j = 0; j < 4; ++j) {
                const uint16_t* bp = brow[j] + k0 + kk * 32 + lh * 8;
                b[j] = as_v16bf(*(const uint4*)bp, *(const uint4*)(bp + 16));
            }
            #pragma unroll
            for (int i = 0; i < 2; ++i)
                #pragma unroll
                for (int j = 0; j < 4; ++j)
                    acc[i][j] = __builtin_amdgcn_wmma_f32_16x16x32_bf16(
                        false, a[i], false, b[j], (short)0, acc[i][j], false, false);
        }
    }

    // D layout: VGPR r -> M = r + 8*lh ; N = lm
    #pragma unroll
    for (int i = 0; i < 2; ++i) {
        const int mrow = mblk + wm * 32 + i * 16 + 8 * lh;
        #pragma unroll
        for (int j = 0; j < 4; ++j) {
            const int n = n0 + j * 16 + lm;
            if (n >= N) continue;
            #pragma unroll
            for (int r = 0; r < 8; ++r) {
                const size_t idx = (size_t)(mrow + r) * N + n;
                float v = acc[i][j][r];
                if (use_resid) v += resid[idx];
                C[idx] = v;
            }
        }
    }
}

// ---------------- 4) causal depthwise conv (K=4) + SiLU ----------------
__global__ __launch_bounds__(256) void conv_silu_kernel(
    const float* __restrict__ zx,     // [T, EPROJ]; xBC at cols [DINNER, DINNER+CONVD)
    const float* __restrict__ cw,     // [CONVD, 4]
    const float* __restrict__ cb,     // [CONVD]
    float* __restrict__ xbc)          // [T, CONVD]
{
    const size_t total = (size_t)TT * CONVD;
    size_t idx = (size_t)blockIdx.x * 256 + threadIdx.x;
    if (idx >= total) return;
    const int c = (int)(idx % CONVD);
    const size_t bs = idx / CONVD;
    const int s = (int)(bs % SS);
    const size_t brow = bs - s;       // b*S
    const float4 w = *(const float4*)(cw + (size_t)c * 4);
    float acc = cb[c];
    const size_t colbase = (size_t)DINNER + c;
    if (s >= 3) acc += zx[(brow + s - 3) * EPROJ + colbase] * w.x;
    if (s >= 2) acc += zx[(brow + s - 2) * EPROJ + colbase] * w.y;
    if (s >= 1) acc += zx[(brow + s - 1) * EPROJ + colbase] * w.z;
    acc += zx[(brow + s) * EPROJ + colbase] * w.w;
    xbc[idx] = acc / (1.f + __expf(-acc));
}

// ---------------- 5) dt = softplus(dt_raw + bias) ----------------
__global__ __launch_bounds__(256) void dt_softplus_kernel(
    const float* __restrict__ zx, const float* __restrict__ dtb,
    float* __restrict__ dt)
{
    const size_t total = (size_t)TT * HEADS;
    size_t idx = (size_t)blockIdx.x * 256 + threadIdx.x;
    if (idx >= total) return;
    const int h = (int)(idx % HEADS);
    const size_t t = idx / HEADS;
    float v = zx[t * EPROJ + (DINNER + CONVD) + h] + dtb[h];
    dt[idx] = (v > 20.f) ? v : log1pf(__expf(v));
}

// ---------------- 6) sequential SSD scan ----------------
// One block per (b, h). Thread t owns state row p = t>>2, cols n0 = (t&3)*32 .. +32.
// y output already includes the D_param * xh skip term.
__global__ __launch_bounds__(256) void ssd_scan_kernel(
    const float* __restrict__ dt,     // [T, HEADS]
    const float* __restrict__ xbc,    // [T, CONVD]: xh [0,2048) Bm [2048,2176) Cm [2176,2304)
    const float* __restrict__ A_log,
    const float* __restrict__ Dp,
    float* __restrict__ y)            // [T, DINNER] == [B,S,H,P]
{
    const int bh = blockIdx.x;
    const int b = bh >> 5;
    const int h = bh & 31;
    const int tid = threadIdx.x;
    const int p  = tid >> 2;
    const int nq = tid & 3;
    const int n0 = nq * 32;
    const float A  = -__expf(A_log[h]);
    const float Dh = Dp[h];

    float hst[32];
    #pragma unroll
    for (int i = 0; i < 32; ++i) hst[i] = 0.f;

    __shared__ float sB[DSTATE];
    __shared__ float sC[DSTATE];

    const size_t brow = (size_t)b * SS;
    for (int s = 0; s < SS; ++s) {
        const size_t row = brow + s;
        const float* xr = xbc + row * CONVD;
        if (tid < 128) sB[tid] = xr[DINNER + tid];
        else           sC[tid - 128] = xr[DINNER + DSTATE + (tid - 128)];
        const float dts = dt[row * HEADS + h];
        const float xt  = xr[h * HDIM + p];
        __syncthreads();
        const float decay = __expf(dts * A);
        const float dx = dts * xt;
        float ysum = 0.f;
        #pragma unroll
        for (int i = 0; i < 32; ++i) {
            float hv = fmaf(hst[i], decay, dx * sB[n0 + i]);
            hst[i] = hv;
            ysum = fmaf(hv, sC[n0 + i], ysum);
        }
        ysum += __shfl_xor(ysum, 1, 32);
        ysum += __shfl_xor(ysum, 2, 32);
        if (nq == 0) y[(row * HEADS + h) * HDIM + p] = ysum + Dh * xt;
        __syncthreads();
    }
}

// ---------------- 7) gate (y * silu(z)) + RMSNorm -> bf16 ----------------
__global__ __launch_bounds__(256) void gate_rmsnorm_to_bf16_kernel(
    const float* __restrict__ y,      // [T, DINNER]
    const float* __restrict__ zx,     // [T, EPROJ]; z at cols [0, DINNER)
    const float* __restrict__ gw,
    uint16_t* __restrict__ out)       // [T, DINNER] bf16
{
    const int t = blockIdx.x;
    const float* yr = y  + (size_t)t * DINNER;
    const float* zr = zx + (size_t)t * EPROJ;
    float vbuf[8];
    float ss = 0.f;
    #pragma unroll
    for (int j = 0; j < 8; ++j) {
        const int i = threadIdx.x + j * 256;
        const float z = zr[i];
        const float v = yr[i] * (z / (1.f + __expf(-z)));
        vbuf[j] = v;
        ss += v * v;
    }
    for (int o = 16; o > 0; o >>= 1) ss += __shfl_xor(ss, o, 32);
    __shared__ float red[8];
    __shared__ float scale_s;
    if ((threadIdx.x & 31) == 0) red[threadIdx.x >> 5] = ss;
    __syncthreads();
    if (threadIdx.x == 0) {
        float s = 0.f;
        #pragma unroll
        for (int i = 0; i < 8; ++i) s += red[i];
        scale_s = rsqrtf(s / (float)DINNER + EPSF);
    }
    __syncthreads();
    const float scale = scale_s;
    #pragma unroll
    for (int j = 0; j < 8; ++j) {
        const int i = threadIdx.x + j * 256;
        out[(size_t)t * DINNER + i] = f2bf(vbuf[j] * scale * gw[i]);
    }
}

// ---------------- launch ----------------
extern "C" void kernel_launch(void* const* d_in, const int* in_sizes, int n_in,
                              void* d_out, int out_size, void* d_ws, size_t ws_size,
                              hipStream_t stream)
{
    const float* x        = (const float*)d_in[0];
    const float* norm_w   = (const float*)d_in[1];
    const float* in_proj  = (const float*)d_in[2];   // [EPROJ, DMODEL]
    const float* conv_w   = (const float*)d_in[3];   // [CONVD, 4]
    const float* conv_b   = (const float*)d_in[4];
    const float* dt_bias  = (const float*)d_in[5];
    const float* A_log    = (const float*)d_in[6];
    const float* D_param  = (const float*)d_in[7];
    const float* gate_w   = (const float*)d_in[8];
    const float* out_proj = (const float*)d_in[9];   // [DMODEL, DINNER]
    float* out = (float*)d_out;

    char* ws = (char*)d_ws;
    size_t off = 0;
    auto take = [&](size_t bytes) -> char* {
        char* p = ws + off;
        off += (bytes + 255) & ~(size_t)255;
        return p;
    };
    uint16_t* xn_bf  = (uint16_t*)take((size_t)TT * DMODEL * 2);
    uint16_t* w1_bf  = (uint16_t*)take((size_t)EPROJ * DMODEL * 2);
    uint16_t* w2_bf  = (uint16_t*)take((size_t)DMODEL * DINNER * 2);
    float*    zx     = (float*)   take((size_t)TT * EPROJ * 4);
    float*    xbc    = (float*)   take((size_t)TT * CONVD * 4);
    float*    dtbuf  = (float*)   take((size_t)TT * HEADS * 4);
    float*    yscan  = (float*)   take((size_t)TT * DINNER * 4);
    uint16_t* y_bf   = (uint16_t*)take((size_t)TT * DINNER * 2);

    // 1) RMSNorm + downcast activations
    rmsnorm_to_bf16_kernel<<<TT, 256, 0, stream>>>(x, norm_w, xn_bf);

    // 2) weight downcasts (L2-resident afterwards)
    {
        size_t n1 = (size_t)EPROJ * DMODEL;
        size_t n2 = (size_t)DMODEL * DINNER;
        f32_to_bf16_kernel<<<2048, 256, 0, stream>>>(in_proj, w1_bf, n1);
        f32_to_bf16_kernel<<<2048, 256, 0, stream>>>(out_proj, w2_bf, n2);
    }

    // 3) GEMM1: zx[T, EPROJ] = xn @ in_proj^T
    {
        dim3 grid(TT / BLK_M, (EPROJ + BLK_N - 1) / BLK_N);
        gemm_bf16_wmma_kernel<<<grid, 256, SMEM_BYTES, stream>>>(
            xn_bf, w1_bf, nullptr, zx, TT, EPROJ, DMODEL, 0);
    }

    // 4) conv + silu
    {
        size_t total = (size_t)TT * CONVD;
        conv_silu_kernel<<<(unsigned)((total + 255) / 256), 256, 0, stream>>>(
            zx, conv_w, conv_b, xbc);
    }

    // 5) dt = softplus
    {
        size_t total = (size_t)TT * HEADS;
        dt_softplus_kernel<<<(unsigned)((total + 255) / 256), 256, 0, stream>>>(
            zx, dt_bias, dtbuf);
    }

    // 6) SSD scan (one block per (b, h))
    ssd_scan_kernel<<<BB * HEADS, 256, 0, stream>>>(dtbuf, xbc, A_log, D_param, yscan);

    // 7) gate + rmsnorm + downcast
    gate_rmsnorm_to_bf16_kernel<<<TT, 256, 0, stream>>>(yscan, zx, gate_w, y_bf);

    // 8) GEMM2: out = residual + y @ out_proj^T
    {
        dim3 grid(TT / BLK_M, DMODEL / BLK_N);
        gemm_bf16_wmma_kernel<<<grid, 256, SMEM_BYTES, stream>>>(
            y_bf, w2_bf, x, out, TT, DMODEL, DINNER, 1);
    }
    (void)in_sizes; (void)n_in; (void)out_size; (void)ws_size;
}
